// MonarchHMM_11278584119746
// MI455X (gfx1250) — matrix operand order repaired and checked by reading, not compiled, not run on hardware
//
#include <hip/hip_runtime.h>
#include <hip/hip_bf16.h>

// ---------------------------------------------------------------------------
// Monarch-HMM backward scan for MI455X (gfx1250, wave32, WMMA).
//
// Decomposition: batch columns are independent through the whole recurrence,
// so each of 8 workgroups owns 8 batch columns and runs the complete
// 255-step scan privately. Per-workgroup state: double-buffered f32 slab
// (8 cols x 4096) in LDS (~256 KB — needs CDNA5's 320 KB LDS/WGP).
// Block matmuls W(64x64) @ exp(X - colmax) use v_wmma_f32_16x16x32_f16.
// W is pre-shuffled into WMMA-A fragment-linear layout so each fragment is
// one coalesced 32 B load per lane; all 8 fragments of a block are loaded
// before the exp/max B-fragment build so LDS+VALU work hides L2 latency.
// ---------------------------------------------------------------------------

#define H_DIM   4096
#define V_DIM   32000
#define BSZ     64
#define NBLK    64              // H / BSZ
#define BATCH   64
#define SEQ     256
#define NCOL    8               // batch columns per workgroup
#define NWG     (BATCH / NCOL)  // 8 workgroups
#define WAVES   16
#define THREADS (WAVES * 32)
#define BPW     (NBLK / WAVES)  // blocks per wave per layer = 4
#define XSTRIDE 4100            // 4096 + 4 pad; %64==4 -> columns hit distinct banks
#define FRAG_PER_LAYER (NBLK * 8 * 32)   // v16h fragments per layer

typedef __attribute__((ext_vector_type(16))) _Float16 v16h;
typedef __attribute__((ext_vector_type(8)))  float    v8f;

__device__ __forceinline__ float fast_exp(float x) {
    // exp(x) = exp2(x * log2(e)); operands here are <= 0 (post max-subtract)
    return __builtin_amdgcn_exp2f(x * 1.44269504088896340736f);
}
__device__ __forceinline__ float fast_log(float x) {
    // log(x) = log2(x) * ln(2); operands are sums of [0,1] values
    return __builtin_amdgcn_logf(x) * 0.69314718055994530942f;
}

// ---------------------------------------------------------------------------
// Init kernel 1: tiled transpose beta (H x V, row-major) -> betaT (V x H)
// so that per-step emission gathers become fully coalesced row reads.
// ---------------------------------------------------------------------------
__global__ void beta_transpose_kernel(const float* __restrict__ beta,
                                      float* __restrict__ betaT) {
    __shared__ float tile[32][33];
    const int v0 = blockIdx.x * 32;
    const int h0 = blockIdx.y * 32;
#pragma unroll
    for (int i = 0; i < 4; ++i) {
        const int h = h0 + threadIdx.y + i * 8;
        tile[threadIdx.y + i * 8][threadIdx.x] =
            beta[(size_t)h * V_DIM + (v0 + threadIdx.x)];
    }
    __syncthreads();
#pragma unroll
    for (int i = 0; i < 4; ++i) {
        const int v = v0 + threadIdx.y + i * 8;
        betaT[(size_t)v * H_DIM + (h0 + threadIdx.x)] =
            tile[threadIdx.x][threadIdx.y + i * 8];
    }
}

// ---------------------------------------------------------------------------
// Init kernel 2: shuffle w0/w1 (f32, 64 blocks of 64x64 row-major) into
// f16 WMMA-A fragment-linear layout:
//   frag[((layer*64+k)*4 + mt)*2 + kc][lane][j]   (16 halves = 32 B per lane)
// A-fragment layout (16-bit A, 16x32): lane row = mt*16 + (lane&15),
// K halves at kc*32 + (lane>>4)*8 + {0..7} and +16.
// One thread per (layer,k,mt,kc,lane) -> each per-lane fragment is one
// contiguous 32 B chunk, so the scan kernel's A loads are fully coalesced.
// ---------------------------------------------------------------------------
__global__ void w_fragment_kernel(const float* __restrict__ w0,
                                  const float* __restrict__ w1,
                                  _Float16* __restrict__ frag) {
    const int idx = blockIdx.x * blockDim.x + threadIdx.x;
    if (idx >= 2 * NBLK * 4 * 2 * 32) return;
    const int lane  = idx & 31;
    const int kc    = (idx >> 5) & 1;
    const int mt    = (idx >> 6) & 3;
    const int k     = (idx >> 8) & 63;
    const int layer = idx >> 14;

    const float* blk = (layer ? w1 : w0) + (size_t)k * (BSZ * BSZ);
    const int row = mt * 16 + (lane & 15);
    const int klo = (lane >> 4) * 8;
    _Float16* dst = frag + (size_t)idx * 16;
#pragma unroll
    for (int j = 0; j < 16; ++j) {
        const int col = kc * 32 + klo + ((j < 8) ? j : (16 + (j - 8)));
        dst[j] = (_Float16)blk[row * BSZ + col];
    }
}

// ---------------------------------------------------------------------------
// One Monarch layer: for each 64-row block k owned by this wave,
//   m[c]   = max_r Xsrc[c][64k + r]
//   E      = exp(Xsrc_blk - m)        (built directly into WMMA B fragments)
//   C      = W_k @ E                  (v_wmma_f32_16x16x32_f16, K-loop of 2)
//   Xdst[c][i*64 + k] = log(C[i][c]) + m[c]   (butterfly permutation)
// Lane mapping: c = lane&15 (only c < NCOL are live), hi = lane>>4.
// ---------------------------------------------------------------------------
__device__ __forceinline__ void monarch_layer(const float* __restrict__ Xsrc,
                                              float* __restrict__ Xdst,
                                              const v16h* __restrict__ wfrag,
                                              int wave, int lane) {
    const int  c   = lane & 15;
    const int  hi  = lane >> 4;        // which 16-lane half
    const bool act = (c < NCOL);

    for (int kb = 0; kb < BPW; ++kb) {
        const int k = wave * BPW + kb;
        const float* xs = Xsrc + c * XSTRIDE + k * BSZ;

        // ---- 1) issue all 8 A-fragment loads for this block (coalesced 32 B
        //         per lane); latency overlaps the max/exp build below.
        v16h areg[8];
#pragma unroll
        for (int f = 0; f < 8; ++f)
            areg[f] = wfrag[(size_t)(k * 8 + f) * 32 + lane];

        // Prefetch next block's fragment region while this one computes.
        if (kb + 1 < BPW)
            __builtin_prefetch(&wfrag[(size_t)(k + 1) * 8 * 32 + lane], 0, 3);

        // ---- 2) per-column max over the 64 contracted rows (lane-pair split)
        float m = -3.0e38f;
        if (act) {
            const float4* xv = (const float4*)(xs + hi * 32);
#pragma unroll
            for (int j = 0; j < 8; ++j) {
                float4 q = xv[j];
                m = fmaxf(m, fmaxf(fmaxf(q.x, q.y), fmaxf(q.z, q.w)));
            }
        }
        m = fmaxf(m, __shfl_xor(m, 16, 32));   // combine lane halves
        if (!act) m = 0.0f;

        // ---- 3) build B fragments on the fly: lane holds E[K=hi*8+j (+16 for j>=8)][N=c]
        v16h bfr[2];
#pragma unroll
        for (int kc = 0; kc < 2; ++kc) {
            v16h b = {};
            if (act) {
                const float4* xq = (const float4*)(xs + kc * 32 + hi * 8);
#pragma unroll
                for (int q4 = 0; q4 < 4; ++q4) {
                    float4 q = xq[q4 + (q4 >> 1) * 2];
                    b[q4 * 4 + 0] = (_Float16)fast_exp(q.x - m);
                    b[q4 * 4 + 1] = (_Float16)fast_exp(q.y - m);
                    b[q4 * 4 + 2] = (_Float16)fast_exp(q.z - m);
                    b[q4 * 4 + 3] = (_Float16)fast_exp(q.w - m);
                }
            }
            bfr[kc] = b;
        }

        // ---- 4) 4 M-tiles of 16 rows, K-loop of 2 per tile
#pragma unroll
        for (int mt = 0; mt < 4; ++mt) {
            v8f cacc = {};
            cacc = __builtin_amdgcn_wmma_f32_16x16x32_f16(
                false, areg[mt * 2 + 0], false, bfr[0], (short)0, cacc,
                false, false);
            cacc = __builtin_amdgcn_wmma_f32_16x16x32_f16(
                false, areg[mt * 2 + 1], false, bfr[1], (short)0, cacc,
                false, false);
            if (act) {
                // C layout: VGPR r -> row mt*16 + r + 8*hi, column c = lane&15.
#pragma unroll
                for (int r = 0; r < 8; ++r) {
                    const int i_local = mt * 16 + r + hi * 8;
                    Xdst[c * XSTRIDE + i_local * BSZ + k] =
                        fast_log(cacc[r]) + m;   // log-domain + butterfly
                }
            }
        }
    }
}

// ---------------------------------------------------------------------------
// Add emission log-probs for timestep t: Xa[c][:] += betaT[ids[b][t]][:].
// Two waves per column, fully coalesced float4 reads of the transposed beta.
// ---------------------------------------------------------------------------
__device__ __forceinline__ void add_emission(float* __restrict__ Xa,
                                             const float* __restrict__ betaT,
                                             const int* __restrict__ ids,
                                             int t, int wg, int wave, int lane) {
    if (wave < NCOL * 2) {
        const int c     = wave >> 1;
        const int hbase = (wave & 1) * (H_DIM / 2);
        const int b     = wg * NCOL + c;
        const int id    = ids[b * SEQ + t];
        const float4* bp = (const float4*)(betaT + (size_t)id * H_DIM + hbase);
        float* xp = Xa + c * XSTRIDE + hbase;
#pragma unroll
        for (int it = 0; it < (H_DIM / 2) / (32 * 4); ++it) {
            const int idx = it * 32 + lane;
            float4 q = bp[idx];
            xp[idx * 4 + 0] += q.x;
            xp[idx * 4 + 1] += q.y;
            xp[idx * 4 + 2] += q.z;
            xp[idx * 4 + 3] += q.w;
        }
    }
}

// ---------------------------------------------------------------------------
// Persistent scan kernel: one workgroup = 8 batch columns, whole sequence.
// ---------------------------------------------------------------------------
__global__ __launch_bounds__(THREADS, 1)
void monarch_hmm_scan_kernel(const float* __restrict__ betaT,
                             const _Float16* __restrict__ wfrag,
                             const int* __restrict__ ids,
                             const float* __restrict__ gamma,
                             float* __restrict__ out) {
    extern __shared__ float smem[];          // 2 * NCOL * XSTRIDE floats
    float* Xa = smem;
    float* Xb = smem + NCOL * XSTRIDE;

    const int tid  = threadIdx.x;
    const int wave = tid >> 5;
    const int lane = tid & 31;
    const int wg   = blockIdx.x;

    // x0 = 0
    for (int i = tid; i < 2 * NCOL * XSTRIDE; i += THREADS) smem[i] = 0.0f;
    __syncthreads();

    const v16h* w0f = (const v16h*)wfrag;                   // applied second
    const v16h* w1f = (const v16h*)wfrag + FRAG_PER_LAYER;  // applied first

    for (int t = SEQ - 1; t >= 1; --t) {
        add_emission(Xa, betaT, ids, t, wg, wave, lane);
        __syncthreads();
        monarch_layer(Xa, Xb, w1f, wave, lane);       // reversed(weights): w1
        __syncthreads();
        monarch_layer(Xb, Xa, w0f, wave, lane);       // then w0
        __syncthreads();
    }

    // t == 0: emission product only, then prior contraction with gamma_exp.
    add_emission(Xa, betaT, ids, 0, wg, wave, lane);
    __syncthreads();

    if (wave < NCOL) {
        const int c = wave;
        const float* xp = Xa + c * XSTRIDE;
        float m = -3.0e38f;
        for (int h = lane; h < H_DIM; h += 32) m = fmaxf(m, xp[h]);
#pragma unroll
        for (int s = 16; s; s >>= 1) m = fmaxf(m, __shfl_xor(m, s, 32));
        float acc = 0.0f;
        for (int h = lane; h < H_DIM; h += 32)
            acc += gamma[h] * fast_exp(xp[h] - m);
#pragma unroll
        for (int s = 16; s; s >>= 1) acc += __shfl_xor(acc, s, 32);
        if (lane == 0) out[wg * NCOL + c] = fast_log(acc) + m;
    }
}

// ---------------------------------------------------------------------------
extern "C" void kernel_launch(void* const* d_in, const int* in_sizes, int n_in,
                              void* d_out, int out_size, void* d_ws,
                              size_t ws_size, hipStream_t stream) {
    (void)in_sizes; (void)n_in; (void)out_size; (void)ws_size;

    const float* w0    = (const float*)d_in[0];   // (64,64,64) f32
    const float* w1    = (const float*)d_in[1];   // (64,64,64) f32
    const float* beta  = (const float*)d_in[2];   // (4096,32000) f32
    const float* gamma = (const float*)d_in[3];   // (1,4096) f32
    const int*   ids   = (const int*)d_in[4];     // (64,256) i32

    float*    betaT = (float*)d_ws;                                   // V*H f32
    _Float16* wfrag = (_Float16*)((char*)d_ws +
                                  (size_t)V_DIM * H_DIM * sizeof(float));

    dim3 tgrid(V_DIM / 32, H_DIM / 32);
    beta_transpose_kernel<<<tgrid, dim3(32, 8), 0, stream>>>(beta, betaT);

    const int nfrag_threads = 2 * NBLK * 4 * 2 * 32;
    w_fragment_kernel<<<(nfrag_threads + 255) / 256, 256, 0, stream>>>(
        w0, w1, wfrag);

    const size_t smem_bytes = (size_t)2 * NCOL * XSTRIDE * sizeof(float);
    monarch_hmm_scan_kernel<<<NWG, THREADS, smem_bytes, stream>>>(
        betaT, wfrag, ids, gamma, (float*)d_out);
}
